// GCN_55018531062479
// MI455X (gfx1250) — matrix operand order repaired and checked by reading, not compiled, not run on hardware
//
#include <hip/hip_runtime.h>

// ---------- CDNA5 WMMA types ----------
typedef __attribute__((ext_vector_type(16))) __bf16 v16bf;
typedef __attribute__((ext_vector_type(8)))  float  v8f;

union V16 {
    v16bf v;
    uint4 q[2];
    unsigned int u[8];
};

__device__ __forceinline__ unsigned short f2bf(float f) {
    unsigned int x = __float_as_uint(f);
    // round-to-nearest-even bf16
    unsigned int r = (x + 0x7FFFu + ((x >> 16) & 1u)) >> 16;
    return (unsigned short)r;
}
__device__ __forceinline__ unsigned int pack2bf(float a, float b) {
    return (unsigned int)f2bf(a) | ((unsigned int)f2bf(b) << 16);
}

// ---------- utility kernels ----------
__global__ void k_zero(float* __restrict__ p, long long n) {
    long long i = (long long)blockIdx.x * blockDim.x + threadIdx.x;
    if (i < n) p[i] = 0.0f;
}

__global__ void k_set_ones(float* __restrict__ p, int n) {
    int i = blockIdx.x * blockDim.x + threadIdx.x;
    if (i < n) p[i] = 1.0f;
}

// deg[dst] += 1 per edge (self-loop counted via init=1.0)
__global__ void k_deg(const long long* __restrict__ ei, float* __restrict__ deg, int E) {
    int e = blockIdx.x * blockDim.x + threadIdx.x;
    if (e < E) atomicAdd(&deg[(int)ei[(long long)E + e]], 1.0f);
}

__global__ void k_rsqrt(float* __restrict__ p, int n) {
    int i = blockIdx.x * blockDim.x + threadIdx.x;
    if (i < n) p[i] = rsqrtf(p[i]);   // deg >= 1 always
}

// f32 -> packed bf16 (2 elems / thread), streaming conversion pass
__global__ void k_cvt_bf16(const float* __restrict__ src, unsigned int* __restrict__ dst,
                           long long n2) {
    long long i = (long long)blockIdx.x * blockDim.x + threadIdx.x;
    if (i < n2) {
        const float2 f = *(const float2*)(src + 2 * i);
        dst[i] = pack2bf(f.x, f.y);
    }
}

// W [K][N] f32 row-major  ->  WT [N][K] bf16 row-major (transpose + convert)
template <int K, int N>
__global__ void k_wtrans(const float* __restrict__ W, unsigned short* __restrict__ WT) {
    int i = blockIdx.x * blockDim.x + threadIdx.x;
    if (i < N * K) {
        int n = i / K, k = i % K;
        WT[i] = f2bf(W[(long long)k * N + n]);
    }
}

// ---------- WMMA GEMM ----------
// C[M x N](f32) = Abf[M x K](bf16) * WT[N x K](bf16)^T
// One wave32 computes a (16*MT) x (16*NT) register tile: MT*NT accumulators,
// each A fragment reused NT times, each B fragment MT times.
// Fragments are two aligned b128 loads each (data pre-packed in WMMA operand order).
// M-tail (odd tile count) handled by clamping the row pointer (duplicate compute,
// wave-uniform store guard) so EXEC stays all-ones at every WMMA.
template <int K, int N, int MT, int NT>
__global__ void k_gemm_wmma(const unsigned short* __restrict__ Abf,
                            const unsigned short* __restrict__ WT,
                            float* __restrict__ C, int n_rows) {
    const int lane   = threadIdx.x & 31;
    const int mtiles = n_rows / 16;
    const int mgrp   = (mtiles + MT - 1) / MT;
    const int ngrp   = N / (16 * NT);
    const int tile   = blockIdx.x * (blockDim.x >> 5) + (threadIdx.x >> 5);
    if (tile >= mgrp * ngrp) return;                 // wave-uniform exit

    const int mt0 = (tile / ngrp) * MT;
    const int ng  = tile % ngrp;
    const int l16 = lane & 15, half = lane >> 4;

    const unsigned short* arow[MT];
#pragma unroll
    for (int mi = 0; mi < MT; ++mi) {
        int mt = mt0 + mi;
        if (mt >= mtiles) mt = mtiles - 1;           // tail clamp (store-guarded below)
        arow[mi] = Abf + (size_t)(mt * 16 + l16) * K;
    }
    const unsigned short* brow[NT];
#pragma unroll
    for (int t = 0; t < NT; ++t)
        brow[t] = WT + (size_t)((ng * NT + t) * 16 + l16) * K;

    v8f acc[MT][NT];
#pragma unroll
    for (int mi = 0; mi < MT; ++mi)
#pragma unroll
        for (int t = 0; t < NT; ++t) acc[mi][t] = {};

    for (int kb = 0; kb < K; kb += 32) {
        V16 a[MT], b[NT];
        // A 16-bit 16x32 operand: dwords = shorts [kb+8*half, +8) and [kb+16+8*half, +8)
#pragma unroll
        for (int mi = 0; mi < MT; ++mi) {
            a[mi].q[0] = *(const uint4*)(arow[mi] + kb + 8 * half);
            a[mi].q[1] = *(const uint4*)(arow[mi] + kb + 16 + 8 * half);
        }
        // B 16-bit 32x16 operand: shorts [kb+8*half, +8) and [kb+16+8*half, +8)
#pragma unroll
        for (int t = 0; t < NT; ++t) {
            b[t].q[0] = *(const uint4*)(brow[t] + kb + 8 * half);
            b[t].q[1] = *(const uint4*)(brow[t] + kb + 16 + 8 * half);
        }
#pragma unroll
        for (int mi = 0; mi < MT; ++mi)
#pragma unroll
            for (int t = 0; t < NT; ++t)
                acc[mi][t] = __builtin_amdgcn_wmma_f32_16x16x32_bf16(
                    false, a[mi].v, false, b[t].v, (short)0, acc[mi][t], false, false);
    }

#pragma unroll
    for (int mi = 0; mi < MT; ++mi) {
        if (mt0 + mi < mtiles) {                     // wave-uniform store guard
#pragma unroll
            for (int t = 0; t < NT; ++t) {
                const int ncol = (ng * NT + t) * 16 + l16;
#pragma unroll
                for (int r = 0; r < 8; ++r) {
                    const int m = (mt0 + mi) * 16 + 8 * half + r;  // C/D: VGPR r -> row 8*half+r
                    C[(size_t)m * N + ncol] = acc[mi][t][r];
                }
            }
        }
    }
}

// ---------- edge propagation: AGG[dst] += H[src] * dis[src]*dis[dst] ----------
template <int C>
__global__ void k_prop(const long long* __restrict__ ei, const float* __restrict__ dis,
                       const float* __restrict__ H, float* __restrict__ AGG, int E) {
    const int CH = C / 4;
    long long gid = (long long)blockIdx.x * blockDim.x + threadIdx.x;
    long long e = gid / CH;
    int c = (int)(gid % CH) * 4;
    if (e >= E) return;
    const int s = (int)ei[e];
    const int d = (int)ei[(long long)E + e];
    const float nrm = dis[s] * dis[d];
    const float4 h = *(const float4*)(H + (size_t)s * C + c);
    float* o = AGG + (size_t)d * C + c;
    atomicAdd(o + 0, h.x * nrm);
    atomicAdd(o + 1, h.y * nrm);
    atomicAdd(o + 2, h.z * nrm);
    atomicAdd(o + 3, h.w * nrm);
}

// H2bf = bf16(relu(AGG1 + H1*dis^2 + b1)); 2 channels/thread, packed dword store
__global__ void k_fin1(const float* __restrict__ AGG, const float* __restrict__ H1,
                       const float* __restrict__ dis, const float* __restrict__ b1,
                       unsigned int* __restrict__ H2bf, int n) {
    int i = blockIdx.x * blockDim.x + threadIdx.x;   // over n*128
    if (i >= n * 128) return;
    const int node = i >> 7;
    const int c = (i & 127) * 2;
    const float dd = dis[node];
    const float s = dd * dd;
    const long long base = (long long)node * 256 + c;
    float v0 = AGG[base]     + H1[base]     * s + b1[c];
    float v1 = AGG[base + 1] + H1[base + 1] * s + b1[c + 1];
    v0 = v0 > 0.0f ? v0 : 0.0f;
    v1 = v1 > 0.0f ? v1 : 0.0f;
    H2bf[i] = pack2bf(v0, v1);
}

// out = out(edge agg) + G*dis^2 + b2
__global__ void k_fin2(float* __restrict__ OUT, const float* __restrict__ G,
                       const float* __restrict__ dis, const float* __restrict__ b2, int n) {
    int i = blockIdx.x * blockDim.x + threadIdx.x;
    if (i >= n * 32) return;
    const int node = i >> 5, c = i & 31;
    const float dd = dis[node];
    OUT[i] = OUT[i] + G[i] * dd * dd + b2[c];
}

// ---------- launch ----------
extern "C" void kernel_launch(void* const* d_in, const int* in_sizes, int n_in,
                              void* d_out, int out_size, void* d_ws, size_t ws_size,
                              hipStream_t stream) {
    const float*     x  = (const float*)d_in[0];
    const long long* ei = (const long long*)d_in[1];
    const float*     W1 = (const float*)d_in[2];
    const float*     b1 = (const float*)d_in[3];
    const float*     W2 = (const float*)d_in[4];
    const float*     b2 = (const float*)d_in[5];
    float*           out = (float*)d_out;

    const int n = in_sizes[0] / 768;   // 50000 (multiple of 16)
    const int E = in_sizes[1] / 2;     // 1600000

    // workspace layout (all offsets 16B aligned)
    char*  ws       = (char*)d_ws;
    size_t xbf_off  = 0;                                   // n*768*2 ; reused as H2bf later
    size_t h1_off   = xbf_off + (size_t)n * 768 * 2;       // n*256*4 ; reused as G2 later
    size_t agg1_off = h1_off + (size_t)n * 256 * 4;        // n*256*4
    size_t dis_off  = agg1_off + (size_t)n * 256 * 4;      // n*4
    size_t w1t_off  = dis_off + (size_t)n * 4;             // 768*256*2
    size_t w2t_off  = w1t_off + (size_t)768 * 256 * 2;     // 256*32*2

    unsigned short* Xbf  = (unsigned short*)(ws + xbf_off);
    unsigned int*   H2bf = (unsigned int*)(ws + xbf_off);  // reuse after gemm1
    float*          H1   = (float*)(ws + h1_off);
    float*          AGG1 = (float*)(ws + agg1_off);
    float*          dis  = (float*)(ws + dis_off);
    unsigned short* W1T  = (unsigned short*)(ws + w1t_off);
    unsigned short* W2T  = (unsigned short*)(ws + w2t_off);
    float*          G2   = H1;                             // reuse after k_fin1

    const int B = 256;
    long long nz1 = (long long)n * 256;
    long long nz2 = (long long)n * 32;

    // zero accumulators
    k_zero<<<(unsigned)((nz1 + B - 1) / B), B, 0, stream>>>(AGG1, nz1);
    k_zero<<<(unsigned)((nz2 + B - 1) / B), B, 0, stream>>>(out, nz2);

    // degrees -> dis = rsqrt(deg)
    k_set_ones<<<(n + B - 1) / B, B, 0, stream>>>(dis, n);
    k_deg<<<(E + B - 1) / B, B, 0, stream>>>(ei, dis, E);
    k_rsqrt<<<(n + B - 1) / B, B, 0, stream>>>(dis, n);

    // precision staging: X -> bf16 (one streaming pass), weights -> bf16 transposed
    long long x2 = (long long)n * 768 / 2;
    k_cvt_bf16<<<(unsigned)((x2 + B - 1) / B), B, 0, stream>>>(x, (unsigned int*)Xbf, x2);
    k_wtrans<768, 256><<<(768 * 256 + B - 1) / B, B, 0, stream>>>(W1, W1T);
    k_wtrans<256, 32><<<(256 * 32 + B - 1) / B, B, 0, stream>>>(W2, W2T);

    // layer 1: H1 = Xbf @ W1 ; wave = 32x64 register tile (8 WMMAs / 12 loads per K-step)
    {
        int mtiles = n / 16;
        int waves = ((mtiles + 1) / 2) * (256 / 64);
        k_gemm_wmma<768, 256, 2, 4><<<(waves + 7) / 8, 256, 0, stream>>>(Xbf, W1T, H1, n);
    }
    long long w1 = (long long)E * 64;
    k_prop<256><<<(unsigned)((w1 + B - 1) / B), B, 0, stream>>>(ei, dis, H1, AGG1, E);
    k_fin1<<<(int)(((long long)n * 128 + B - 1) / B), B, 0, stream>>>(AGG1, H1, dis, b1, H2bf, n);

    // layer 2: G2 = H2bf @ W2 ; wave = 32x32 register tile
    {
        int mtiles = n / 16;
        int waves = ((mtiles + 1) / 2) * (32 / 32);
        k_gemm_wmma<256, 32, 2, 2><<<(waves + 7) / 8, 256, 0, stream>>>((unsigned short*)H2bf, W2T, G2, n);
    }
    long long w2 = (long long)E * 8;
    k_prop<32><<<(unsigned)((w2 + B - 1) / B), B, 0, stream>>>(ei, dis, G2, out, E);
    k_fin2<<<(int)((nz2 + B - 1) / B), B, 0, stream>>>(out, G2, dis, b2, n);
}
